// SSD_two_branch_54125177865034
// MI455X (gfx1250) — matrix operand order repaired and checked by reading, not compiled, not run on hardware
//
#include <hip/hip_runtime.h>
#include <hip/hip_bf16.h>

// ---------------------------------------------------------------------------
// SSD300 two-branch forward for gfx1250 (MI455X).
// Strategy: implicit-GEMM convolutions on V_WMMA_F32_16X16X32_BF16.
//  - Activations: NHWC bf16 (all trunk channels are multiples of 32).
//  - K ordering: tap-major, channel-minor -> each 32-K chunk = 32 contiguous
//    channels -> A fragment is two 16B loads per lane (documented layout).
//  - Weights: packed per-call into per-lane-contiguous B fragments (32B/lane).
//  - One wave computes a 16px x 64ch tile: 1 A-frag feeds 4 WMMAs.
//  - conv1_1 (C=3) is scalar (1% of FLOPs); heads write fp32 directly into
//    the flat_head layout of d_out; priors computed on device.
// ---------------------------------------------------------------------------

typedef __bf16 bf16;
typedef __attribute__((ext_vector_type(16))) __bf16        v16bf;
typedef __attribute__((ext_vector_type(8)))  float         v8f;
typedef __attribute__((ext_vector_type(8)))  unsigned int  v8u;
typedef __attribute__((ext_vector_type(4)))  unsigned int  v4u;

// ---------------- weight packing: OIHW fp32 -> B-fragment bf16 -------------
// Layout: dword index d = (((t*nchunks + kc)*32 + lane)*8 + v)
// Lane (n = lane&15, half = lane>>4) dword v holds K pair k0,k0+1 with
// k0 = kc*32 + 2*(v&3) + 8*half + 16*(v>>2)  (dense 16x16x32 bf16 B layout).
__global__ void pack_weights(const float* __restrict__ w, unsigned int* __restrict__ wpk,
                             int Cout, int C, int FS, int ntiles, int nchunks)
{
    long d = (long)blockIdx.x * blockDim.x + threadIdx.x;
    long total = (long)ntiles * nchunks * 32 * 8;
    if (d >= total) return;
    int v    = (int)(d & 7);
    int lane = (int)((d >> 3) & 31);
    long rest = d >> 8;
    int kc = (int)(rest % nchunks);
    int t  = (int)(rest / nchunks);
    int half = lane >> 4, n = lane & 15;
    int o = t * 16 + n;
    int Ktot = C * FS * FS;
    unsigned int pk = 0;
#pragma unroll
    for (int j = 0; j < 2; ++j) {
        int k = kc * 32 + 2 * (v & 3) + 8 * half + 16 * (v >> 2) + j;
        float val = 0.0f;
        if (o < Cout && k < Ktot) {
            int tap = k / C, c = k % C;
            int ry = tap / FS, rx = tap - ry * FS;
            val = w[(((long)o * C + c) * FS + ry) * FS + rx];
        }
        unsigned short bits = __builtin_bit_cast(unsigned short, (bf16)val);
        pk |= (unsigned int)bits << (16 * j);
    }
    wpk[d] = pk;
}

// ---------------- generic implicit-GEMM conv (bf16 WMMA) -------------------
__global__ __launch_bounds__(32) void conv_wmma(
    const bf16* __restrict__ in, const unsigned int* __restrict__ wpk,
    const float* __restrict__ bias,
    bf16* __restrict__ out_bf, float* __restrict__ out_f32,
    int H, int Wd, int C, int OH, int OW, int Cout,
    int FS, int stride, int pad, int dil,
    long long out_boff, long long out_bstride)
{
    const int lane = threadIdx.x;
    const int half = lane >> 4;
    const int m16  = lane & 15;
    const int ptile = blockIdx.x;
    const int b     = blockIdx.z;
    const int OHOW  = OH * OW;
    const int ntiles  = (Cout + 15) >> 4;
    const int nchunks = (C * FS * FS) >> 5;
    int t0 = blockIdx.y * 4;
    int nt = ntiles - t0; if (nt > 4) nt = 4;
    const long tstride = (long)nchunks * 32 * 8;

    // A-fragment row for this lane
    int pA = ptile * 16 + m16;
    bool prow = pA < OHOW;
    int pa = prow ? pA : 0;
    int oy = pa / OW, ox = pa - oy * OW;
    const bf16* inb = in + (long)b * H * Wd * C;

    v8f acc[4] = {v8f{}, v8f{}, v8f{}, v8f{}};

    int kc = 0;
    for (int tap = 0; tap < FS * FS; ++tap) {
        int ry = tap / FS, rx = tap - ry * FS;
        int iy = oy * stride - pad + ry * dil;
        int ix = ox * stride - pad + rx * dil;
        bool ok = prow && (iy >= 0) && (iy < H) && (ix >= 0) && (ix < Wd);
        const bf16* ap = inb + ((long)iy * Wd + ix) * C + 8 * half;
        for (int c0 = 0; c0 < C; c0 += 32, ++kc) {
            v4u alo = {0u, 0u, 0u, 0u}, ahi = {0u, 0u, 0u, 0u};
            if (ok) {
                alo = *reinterpret_cast<const v4u*>(ap + c0);
                ahi = *reinterpret_cast<const v4u*>(ap + c0 + 16);
            }
            v8u a8;
            a8[0] = alo[0]; a8[1] = alo[1]; a8[2] = alo[2]; a8[3] = alo[3];
            a8[4] = ahi[0]; a8[5] = ahi[1]; a8[6] = ahi[2]; a8[7] = ahi[3];
            v16bf af = __builtin_bit_cast(v16bf, a8);

            const unsigned int* wl = wpk + (((long)t0 * nchunks + kc) * 32 + lane) * 8;
            __builtin_prefetch((const void*)(wl + 256), 0, 1); // next K chunk, tile0
#pragma unroll
            for (int t = 0; t < 4; ++t) {
                if (t < nt) {
                    v8u bw = *reinterpret_cast<const v8u*>(wl + (long)t * tstride);
                    v16bf bfm = __builtin_bit_cast(v16bf, bw);
                    acc[t] = __builtin_amdgcn_wmma_f32_16x16x32_bf16(
                        false, af, false, bfm, (short)0, acc[t], false, false);
                }
            }
        }
    }

    // Epilogue: C/D layout -> lane n = m16 is out-channel, VGPR v -> row v+8*half
#pragma unroll
    for (int t = 0; t < 4; ++t) {
        if (t >= nt) break;
        int ch = (t0 + t) * 16 + m16;
        if (ch < Cout) {
            float bv = bias[ch];
#pragma unroll
            for (int v = 0; v < 8; ++v) {
                int p = ptile * 16 + v + 8 * half;
                if (p < OHOW) {
                    float val = acc[t][v] + bv;
                    if (out_f32) {
                        out_f32[(long)b * out_bstride + out_boff + (long)p * Cout + ch] = val;
                    } else {
                        out_bf[((long)b * OHOW + p) * Cout + ch] = (bf16)(val > 0.0f ? val : 0.0f);
                    }
                }
            }
        }
    }
}

// ---------------- conv1_1: 3->64 @300x300, scalar (C%32 != 0) --------------
__global__ void conv1_direct(const float* __restrict__ x, const float* __restrict__ w,
                             const float* __restrict__ bias, bf16* __restrict__ out)
{
    long idx = (long)blockIdx.x * blockDim.x + threadIdx.x;
    const long total = 8L * 90000 * 64;
    if (idx >= total) return;
    int oc = (int)(idx & 63);
    long r = idx >> 6;
    int p = (int)(r % 90000);
    int b = (int)(r / 90000);
    int oy = p / 300, ox = p - oy * 300;
    float acc = bias[oc];
#pragma unroll
    for (int c = 0; c < 3; ++c)
        for (int ry = 0; ry < 3; ++ry) {
            int iy = oy - 1 + ry;
            if (iy < 0 || iy >= 300) continue;
            for (int rx = 0; rx < 3; ++rx) {
                int ix = ox - 1 + rx;
                if (ix < 0 || ix >= 300) continue;
                acc += x[(((long)b * 3 + c) * 300 + iy) * 300 + ix] *
                       w[((oc * 3 + c) * 3 + ry) * 3 + rx];
            }
        }
    out[idx] = (bf16)(acc > 0.0f ? acc : 0.0f); // idx == ((b*90000+p)*64+oc)
}

// ---------------- maxpool, NHWC bf16, OOB = -inf (jax reduce_window) -------
__global__ void maxpool2d(const bf16* __restrict__ in, bf16* __restrict__ out,
                          int H, int Wd, int C, int OH, int OW, int ks, int st, int padt)
{
    long idx = (long)blockIdx.x * blockDim.x + threadIdx.x;
    long total = 8L * OH * OW * C;
    if (idx >= total) return;
    int c = (int)(idx % C);
    long r = idx / C;
    int p = (int)(r % (OH * OW));
    int b = (int)(r / (OH * OW));
    int oy = p / OW, ox = p - oy * OW;
    float m = -3.4e38f;
    for (int ky = 0; ky < ks; ++ky) {
        int iy = oy * st - padt + ky;
        if (iy < 0 || iy >= H) continue;
        for (int kx = 0; kx < ks; ++kx) {
            int ix = ox * st - padt + kx;
            if (ix < 0 || ix >= Wd) continue;
            float v = (float)in[(((long)b * H + iy) * Wd + ix) * C + c];
            m = v > m ? v : m;
        }
    }
    out[(((long)b * OH + oy) * OW + ox) * C + c] = (bf16)m;
}

// ---------------- L2Norm over 512 channels (conv4_3 source) ----------------
__global__ void l2norm_k(const bf16* __restrict__ in, const float* __restrict__ w,
                         bf16* __restrict__ out)
{
    int p = blockIdx.x * blockDim.x + threadIdx.x;
    if (p >= 8 * 38 * 38) return;
    const bf16* ip = in + (long)p * 512;
    float s = 0.0f;
    for (int c = 0; c < 512; ++c) { float v = (float)ip[c]; s += v * v; }
    float inv = 1.0f / (sqrtf(s) + 1e-10f);
    bf16* op = out + (long)p * 512;
    for (int c = 0; c < 512; ++c) op[c] = (bf16)((float)ip[c] * inv * w[c]);
}

// ---------------- SSD PriorBox (written into both output slots) ------------
__device__ __forceinline__ float clamp01(float x) { return x < 0.f ? 0.f : (x > 1.f ? 1.f : x); }

__global__ void priors_kernel(float* __restrict__ out1, float* __restrict__ out2)
{
    int idx = blockIdx.x * blockDim.x + threadIdx.x;
    if (idx >= 8732) return;
    const int off[7]    = {0, 5776, 7942, 8542, 8692, 8728, 8732};
    const int fm[6]     = {38, 19, 10, 5, 3, 1};
    const int mb[6]     = {4, 6, 6, 6, 4, 4};
    const float stepv[6] = {8.f, 16.f, 32.f, 64.f, 100.f, 300.f};
    const float smin[6] = {30.f/300.f, 60.f/300.f, 111.f/300.f, 162.f/300.f, 213.f/300.f, 264.f/300.f};
    const float smax[6] = {60.f/300.f, 111.f/300.f, 162.f/300.f, 213.f/300.f, 264.f/300.f, 315.f/300.f};
    int k = 0;
    while (k < 5 && idx >= off[k + 1]) ++k;
    int bi = idx - off[k];
    int m = mb[k];
    int cell = bi / m, v = bi - cell * m;
    int f = fm[k];
    int i = cell / f, j = cell - i * f;
    float fk = 300.0f / stepv[k];
    float cx = (j + 0.5f) / fk, cy = (i + 0.5f) / fk;
    float s = smin[k], sp = sqrtf(s * smax[k]);
    float w, h;
    if (v == 0)      { w = s;  h = s;  }
    else if (v == 1) { w = sp; h = sp; }
    else {
        int a = (v - 2) >> 1, flip = (v - 2) & 1;
        float ar = (a == 0) ? 2.0f : 3.0f;
        float rr = sqrtf(ar);
        if (!flip) { w = s * rr; h = s / rr; } else { w = s / rr; h = s * rr; }
    }
    float4 pv = make_float4(clamp01(cx), clamp01(cy), clamp01(w), clamp01(h));
    reinterpret_cast<float4*>(out1)[idx] = pv;
    reinterpret_cast<float4*>(out2)[idx] = pv;
}

// ---------------------------------------------------------------------------
extern "C" void kernel_launch(void* const* d_in, const int* in_sizes, int n_in,
                              void* d_out, int out_size, void* d_ws, size_t ws_size,
                              hipStream_t stream)
{
    (void)in_sizes; (void)n_in; (void)out_size; (void)ws_size;
    const float* x = (const float*)d_in[0];
    auto VW = [&](int i) { return (const float*)d_in[1 + 2 * i]; };
    auto VB = [&](int i) { return (const float*)d_in[2 + 2 * i]; };
    const float* l2w = (const float*)d_in[31];
    auto EW = [&](int i) { return (const float*)d_in[32 + 2 * i]; };
    auto EB = [&](int i) { return (const float*)d_in[33 + 2 * i]; };
    auto HWp = [&](int base, int j) { return (const float*)d_in[base + 2 * j]; };
    auto HBp = [&](int base, int j) { return (const float*)d_in[base + 2 * j + 1]; };

    char* ws = (char*)d_ws;
    size_t off = 0;
    auto balloc = [&](size_t bytes) -> char* {
        char* p = ws + off;
        off = (off + bytes + 255) & ~(size_t)255;
        return p;
    };
    bf16* A  = (bf16*)balloc(92160000);   // 8*300*300*64 bf16 ping
    bf16* Bb = (bf16*)balloc(92160000);   // pong
    const int fm[6]  = {38, 19, 10, 5, 3, 1};
    const int sch[6] = {512, 1024, 512, 256, 256, 256};
    bf16* SRC[6];
    for (int s = 0; s < 6; ++s) SRC[s] = (bf16*)balloc((size_t)8 * fm[s] * fm[s] * sch[s] * 2);
    size_t woff = off;  // packed-weight bump allocator

    float* out = (float*)d_out;

    auto conv = [&](const bf16* in, int H, int Wd, int C, int Cout, int FS, int st, int pad, int dil,
                    const float* w, const float* bias,
                    bf16* obf, float* of32, long long boff, long long bstride) {
        int OH = (H  + 2 * pad - dil * (FS - 1) - 1) / st + 1;
        int OW = (Wd + 2 * pad - dil * (FS - 1) - 1) / st + 1;
        int nchunks = (C * FS * FS) >> 5;
        int ntiles  = (Cout + 15) >> 4;
        unsigned int* wpk = (unsigned int*)(ws + woff);
        size_t nwords = (size_t)ntiles * nchunks * 32 * 8;
        woff = (woff + nwords * 4 + 255) & ~(size_t)255;
        pack_weights<<<dim3((unsigned)((nwords + 255) / 256)), dim3(256), 0, stream>>>(
            w, wpk, Cout, C, FS, ntiles, nchunks);
        dim3 g((OH * OW + 15) / 16, (ntiles + 3) / 4, 8);
        conv_wmma<<<g, dim3(32), 0, stream>>>(in, wpk, bias, obf, of32,
                                              H, Wd, C, OH, OW, Cout, FS, st, pad, dil,
                                              boff, bstride);
    };
    auto pool = [&](const bf16* in, bf16* o, int H, int Wd, int C, int OH, int OW,
                    int ks, int st, int padt) {
        long total = 8L * OH * OW * C;
        maxpool2d<<<dim3((unsigned)((total + 255) / 256)), dim3(256), 0, stream>>>(
            in, o, H, Wd, C, OH, OW, ks, st, padt);
    };

    // ---- VGG trunk ----
    {
        long total = 8L * 90000 * 64;
        conv1_direct<<<dim3((unsigned)((total + 255) / 256)), dim3(256), 0, stream>>>(x, VW(0), VB(0), A);
    }
    conv(A,  300, 300,  64,  64, 3, 1, 1, 1, VW(1),  VB(1),  Bb, nullptr, 0, 0);   // conv1_2
    pool(Bb, A, 300, 300, 64, 150, 150, 2, 2, 0);
    conv(A,  150, 150,  64, 128, 3, 1, 1, 1, VW(2),  VB(2),  Bb, nullptr, 0, 0);   // conv2_1
    conv(Bb, 150, 150, 128, 128, 3, 1, 1, 1, VW(3),  VB(3),  A,  nullptr, 0, 0);   // conv2_2
    pool(A, Bb, 150, 150, 128, 75, 75, 2, 2, 0);
    conv(Bb,  75,  75, 128, 256, 3, 1, 1, 1, VW(4),  VB(4),  A,  nullptr, 0, 0);   // conv3_1
    conv(A,   75,  75, 256, 256, 3, 1, 1, 1, VW(5),  VB(5),  Bb, nullptr, 0, 0);   // conv3_2
    conv(Bb,  75,  75, 256, 256, 3, 1, 1, 1, VW(6),  VB(6),  A,  nullptr, 0, 0);   // conv3_3
    pool(A, Bb, 75, 75, 256, 38, 38, 2, 2, 0);                                     // ceil-mode
    conv(Bb,  38,  38, 256, 512, 3, 1, 1, 1, VW(7),  VB(7),  A,  nullptr, 0, 0);   // conv4_1
    conv(A,   38,  38, 512, 512, 3, 1, 1, 1, VW(8),  VB(8),  Bb, nullptr, 0, 0);   // conv4_2
    conv(Bb,  38,  38, 512, 512, 3, 1, 1, 1, VW(9),  VB(9),  A,  nullptr, 0, 0);   // conv4_3
    {
        int np = 8 * 38 * 38;
        l2norm_k<<<dim3((np + 255) / 256), dim3(256), 0, stream>>>(A, l2w, SRC[0]);
    }
    pool(A, Bb, 38, 38, 512, 19, 19, 2, 2, 0);
    conv(Bb,  19,  19, 512, 512, 3, 1, 1, 1, VW(10), VB(10), A,  nullptr, 0, 0);   // conv5_1
    conv(A,   19,  19, 512, 512, 3, 1, 1, 1, VW(11), VB(11), Bb, nullptr, 0, 0);   // conv5_2
    conv(Bb,  19,  19, 512, 512, 3, 1, 1, 1, VW(12), VB(12), A,  nullptr, 0, 0);   // conv5_3
    pool(A, Bb, 19, 19, 512, 19, 19, 3, 1, 1);                                     // pool5 k3s1p1
    conv(Bb,  19,  19, 512, 1024, 3, 1, 6, 6, VW(13), VB(13), A,      nullptr, 0, 0); // conv6 dil6
    conv(A,   19,  19, 1024, 1024, 1, 1, 0, 1, VW(14), VB(14), SRC[1], nullptr, 0, 0); // conv7

    // ---- extras ----
    conv(SRC[1], 19, 19, 1024, 256, 1, 1, 0, 1, EW(0), EB(0), A,      nullptr, 0, 0);
    conv(A,      19, 19,  256, 512, 3, 2, 1, 1, EW(1), EB(1), SRC[2], nullptr, 0, 0); // 10x10
    conv(SRC[2], 10, 10,  512, 128, 1, 1, 0, 1, EW(2), EB(2), A,      nullptr, 0, 0);
    conv(A,      10, 10,  128, 256, 3, 2, 1, 1, EW(3), EB(3), SRC[3], nullptr, 0, 0); // 5x5
    conv(SRC[3],  5,  5,  256, 128, 1, 1, 0, 1, EW(4), EB(4), A,      nullptr, 0, 0);
    conv(A,       5,  5,  128, 256, 3, 1, 0, 1, EW(5), EB(5), SRC[4], nullptr, 0, 0); // 3x3
    conv(SRC[4],  3,  3,  256, 128, 1, 1, 0, 1, EW(6), EB(6), A,      nullptr, 0, 0);
    conv(A,       3,  3,  128, 256, 3, 1, 0, 1, EW(7), EB(7), SRC[5], nullptr, 0, 0); // 1x1

    // ---- heads: write fp32 directly into flat_head layout of d_out ----
    const int  locCh[6]   = {16, 24, 24, 24, 16, 16};
    const int  confCh[6]  = {8, 12, 12, 12, 8, 8};
    const long locOff[6]  = {0, 23104, 31768, 34168, 34768, 34912};
    const long confOff[6] = {0, 11552, 15884, 17084, 17384, 17456};
    float* carLoc  = out;            // [8, 8732, 4], batch stride 34928
    float* carConf = out + 279424;   // [8, 8732, 2], batch stride 17464
    float* cpLoc   = out + 454064;
    float* cpConf  = out + 733488;
    for (int s = 0; s < 6; ++s) {
        int f = fm[s], C = sch[s];
        conv(SRC[s], f, f, C, locCh[s],  3, 1, 1, 1, HWp(48, s), HBp(48, s), nullptr, carLoc,  locOff[s],  34928);
        conv(SRC[s], f, f, C, confCh[s], 3, 1, 1, 1, HWp(60, s), HBp(60, s), nullptr, carConf, confOff[s], 17464);
        conv(SRC[s], f, f, C, locCh[s],  3, 1, 1, 1, HWp(72, s), HBp(72, s), nullptr, cpLoc,   locOff[s],  34928);
        conv(SRC[s], f, f, C, confCh[s], 3, 1, 1, 1, HWp(84, s), HBp(84, s), nullptr, cpConf,  confOff[s], 17464);
    }

    // ---- priors (both branches share them) ----
    priors_kernel<<<dim3((8732 + 255) / 256), dim3(256), 0, stream>>>(out + 419136, out + 873200);
}
